// LSTM_64914135711810
// MI455X (gfx1250) — matrix operand order repaired and checked by reading, not compiled, not run on hardware
//
#include <hip/hip_runtime.h>
#include <cstddef>

typedef __bf16 bf16_t;
typedef __bf16 v16bf __attribute__((ext_vector_type(16)));
typedef __bf16 v8bf  __attribute__((ext_vector_type(8)));
typedef float  v8f   __attribute__((ext_vector_type(8)));

#define T_STEPS 512
#define BATCH   64
#define EDIM    1024
#define HDIM    1024

// gfx1250 native transcendental tanh (V_TANH_F32)
__device__ __forceinline__ float hw_tanh(float x) {
  float y;
  asm("v_tanh_f32 %0, %1" : "=v"(y) : "v"(x));
  return y;
}

// ---------------------------------------------------------------------------
// One-time per-launch fp32 -> bf16 conversion of both weight matrices.
// ---------------------------------------------------------------------------
__global__ void lstm_cvt_weights(const float* __restrict__ wx,
                                 const float* __restrict__ wh,
                                 bf16_t* __restrict__ wxb,
                                 bf16_t* __restrict__ whb, int n) {
  int i = blockIdx.x * blockDim.x + threadIdx.x;
  if (i < n) {
    wxb[i] = (bf16_t)wx[i];
    whb[i] = (bf16_t)wh[i];
  }
}

// ---------------------------------------------------------------------------
// Zero h0/h1 (bf16 ping-pong) and c (fp32) each launch (deterministic).
// ---------------------------------------------------------------------------
__global__ void lstm_init_state(float* __restrict__ c,
                                bf16_t* __restrict__ h0,
                                bf16_t* __restrict__ h1, int n) {
  int i = blockIdx.x * blockDim.x + threadIdx.x;
  if (i < n) {
    c[i]  = 0.0f;
    h0[i] = (bf16_t)0.0f;
    h1[i] = (bf16_t)0.0f;
  }
}

// ---------------------------------------------------------------------------
// One LSTM timestep. Grid: (HDIM/16 = 64). Block: 128 threads = 4 waves.
// Block covers ALL 64 batch rows for one 16-wide hidden-column tile; wave g
// computes gate g with 4 v8f accumulators (batch tiles). A slabs (64x32) are
// double-buffered in LDS (one barrier per k-step); phase 2 stages h via
// GLOBAL_LOAD_ASYNC_TO_LDS_B128 overlapped with the current slab's WMMAs.
// ---------------------------------------------------------------------------
__global__ __launch_bounds__(128) void lstm_step(
    const float*  __restrict__ x_t,    // [B, E]   embeds + t*B*E (fp32)
    const bf16_t* __restrict__ wxb,    // [4H, E]  bf16, row-major (K contiguous)
    const bf16_t* __restrict__ whb,    // [4H, H]  bf16, row-major (K contiguous)
    const bf16_t* __restrict__ hprev,  // [B, H]   bf16
    const float*  __restrict__ bi, const float* __restrict__ bff,
    const float*  __restrict__ bo, const float* __restrict__ bg,
    float*  __restrict__ cst,          // [B, H]   fp32 cell state (in/out)
    bf16_t* __restrict__ hnext,        // [B, H]   bf16 (out, other ping buffer)
    float*  __restrict__ out_t)        // [B, H]   fp32 = out + t*B*H
{
  __shared__ __align__(16) bf16_t Abuf[2][BATCH * 32];  // 2 x 4KB ping-pong
  __shared__ float zs[4][BATCH][16];                    // 16KB z tiles

  const int tid  = threadIdx.x;
  const int gate = tid >> 5;           // wave id = gate 0..3  [i|f|o|g]
  const int lane = tid & 31;
  const int row    = lane & 15;        // A-matrix: M = lane%16 (within tile)
  const int ncol   = lane & 15;        // B/C/D: N = lane%16
  const int khalfA = (lane >> 4) * 16; // A k-offset: lanes 16-31 hold K=16..31
  const int kbaseB = (lane >> 4) * 16; // B k-offset: lanes 16-31 hold K=16..31
  const int htile  = blockIdx.x << 4;

  // Bias folded into accumulator init: every C element in a lane is column
  // (htile+ncol), so all 8 start at bias[col].
  const float* bias = (gate == 0) ? bi : (gate == 1) ? bff : (gate == 2) ? bo : bg;
  const float bv    = bias[htile + ncol];
  v8f acc[4];
#pragma unroll
  for (int mt = 0; mt < 4; ++mt)
#pragma unroll
    for (int r = 0; r < 8; ++r) acc[mt][r] = bv;

  // ========== Phase 1: x contribution (stage+convert, ping-pong LDS) =======
  const bf16_t* wrow_x = wxb + ((size_t)gate * HDIM + htile + ncol) * EDIM;
  for (int i = 0; i < EDIM / 32; ++i) {
    const int k0 = i * 32;
    bf16_t* dst = Abuf[i & 1];
    // Stage: 256 chunks of 8 elements; 128 threads x 2 chunks; f32 -> bf16.
#pragma unroll
    for (int e = 0; e < 2; ++e) {
      const int q   = tid + e * 128;   // chunk id 0..255
      const int r   = q >> 2;          // batch row 0..63
      const int sub = q & 3;           // 8-element group within 32-k slab
      const float4* src = (const float4*)(x_t + (size_t)r * EDIM + k0 + sub * 8);
      const float4 fa = src[0];
      const float4 fb = src[1];
      v8bf o;
      o[0] = (bf16_t)fa.x; o[1] = (bf16_t)fa.y; o[2] = (bf16_t)fa.z; o[3] = (bf16_t)fa.w;
      o[4] = (bf16_t)fb.x; o[5] = (bf16_t)fb.y; o[6] = (bf16_t)fb.z; o[7] = (bf16_t)fb.w;
      *(v8bf*)(dst + (size_t)q * 8) = o;
    }
    __syncthreads();                   // slab i visible; prior readers done

    const v16bf b = *(const v16bf*)(wrow_x + k0 + kbaseB);  // 32B contiguous
    __builtin_prefetch(wrow_x + k0 + 32 + kbaseB, 0, 1);    // global_prefetch_b8
#pragma unroll
    for (int mt = 0; mt < 4; ++mt) {
      const bf16_t* ap = dst + ((size_t)(mt * 16 + row)) * 32 + khalfA;
      const v8bf lo = *(const v8bf*)(ap);        // ds_load_b128
      const v8bf hi = *(const v8bf*)(ap + 8);    // ds_load_b128
      const v16bf a = __builtin_shufflevector(lo, hi,
          0,1,2,3,4,5,6,7,8,9,10,11,12,13,14,15);
      acc[mt] = __builtin_amdgcn_wmma_f32_16x16x32_bf16(
          false, a, false, b, (short)0, acc[mt], false, false);
    }
  }

  // ========== Phase 2: h contribution (async-to-LDS pipeline) ==============
  const bf16_t* wrow_h = whb + ((size_t)gate * HDIM + htile + ncol) * HDIM;

  // Prologue: stage slab 0 into buffer 0 (safe: phase-1 barrier chain orders
  // this after all reads of Abuf[0]).
#pragma unroll
  for (int e = 0; e < 2; ++e) {
    const int q = tid + e * 128;
    const bf16_t* gp = hprev + (size_t)(q >> 2) * HDIM + (q & 3) * 8;
    const unsigned ldsoff = (unsigned)(size_t)(Abuf[0] + (size_t)q * 8);
    asm volatile("global_load_async_to_lds_b128 %0, %1, off"
                 :: "v"(ldsoff), "v"(gp) : "memory");
  }

  for (int i = 0; i < HDIM / 32; ++i) {
    const int k0 = i * 32;
    asm volatile("s_wait_asynccnt 0" ::: "memory");  // my slab-i loads done
    __syncthreads();                                 // everyone's done; buf ready

    // Issue async stage of slab i+1 into the other buffer; overlaps compute.
    if (i + 1 < HDIM / 32) {
#pragma unroll
      for (int e = 0; e < 2; ++e) {
        const int q = tid + e * 128;
        const bf16_t* gp = hprev + (size_t)(q >> 2) * HDIM + (k0 + 32) + (q & 3) * 8;
        const unsigned ldsoff =
            (unsigned)(size_t)(Abuf[(i + 1) & 1] + (size_t)q * 8);
        asm volatile("global_load_async_to_lds_b128 %0, %1, off"
                     :: "v"(ldsoff), "v"(gp) : "memory");
      }
    }

    const bf16_t* abase = Abuf[i & 1];
    const v16bf b = *(const v16bf*)(wrow_h + k0 + kbaseB);
    __builtin_prefetch(wrow_h + k0 + 32 + kbaseB, 0, 1);
#pragma unroll
    for (int mt = 0; mt < 4; ++mt) {
      const bf16_t* ap = abase + ((size_t)(mt * 16 + row)) * 32 + khalfA;
      const v8bf lo = *(const v8bf*)(ap);
      const v8bf hi = *(const v8bf*)(ap + 8);
      const v16bf a = __builtin_shufflevector(lo, hi,
          0,1,2,3,4,5,6,7,8,9,10,11,12,13,14,15);
      acc[mt] = __builtin_amdgcn_wmma_f32_16x16x32_bf16(
          false, a, false, b, (short)0, acc[mt], false, false);
    }
  }

  // ---- stash z tiles in LDS (C/D layout: VGPR r -> M = r + (lane/16)*8) ---
  const int mbase = (lane >> 4) * 8;
#pragma unroll
  for (int mt = 0; mt < 4; ++mt)
#pragma unroll
    for (int r = 0; r < 8; ++r)
      zs[gate][mt * 16 + mbase + r][ncol] = acc[mt][r];

  __syncthreads();

  // ---- elementwise gate math: 64x16 = 1024 elements across 128 threads ----
#pragma unroll
  for (int e = 0; e < 8; ++e) {
    const int idx = tid + e * 128;
    const int m = idx >> 4, n = idx & 15;  // m = batch row, n = col in tile
    const float zi = zs[0][m][n];
    const float zf = zs[1][m][n];
    const float zo = zs[2][m][n];
    const float zg = zs[3][m][n];
    const float ig = 1.0f / (1.0f + __expf(-zi));
    const float fg = 1.0f / (1.0f + __expf(-zf));
    const float og = 1.0f / (1.0f + __expf(-zo));
    const float gg = hw_tanh(zg);            // v_tanh_f32
    const size_t off = (size_t)m * HDIM + (htile + n);
    const float cn = gg * ig + cst[off] * fg;
    const float hn = hw_tanh(cn) * og;       // v_tanh_f32
    cst[off]   = cn;
    hnext[off] = (bf16_t)hn;
    out_t[off] = hn;
  }
}

// ---------------------------------------------------------------------------
// Launch: convert weights, zero state, then 512 dependent step kernels.
// Workspace layout (~16.6 MB):
//   [0, 8MB)        w_x bf16   [4096 x 1024]
//   [8MB, 16MB)     w_h bf16   [4096 x 1024]
//   [16MB, +256KB)  c fp32     [64 x 1024]
//   next 128KB      h ping 0   [64 x 1024] bf16
//   next 128KB      h ping 1   [64 x 1024] bf16
// ---------------------------------------------------------------------------
extern "C" void kernel_launch(void* const* d_in, const int* in_sizes, int n_in,
                              void* d_out, int out_size, void* d_ws, size_t ws_size,
                              hipStream_t stream) {
  const float* embeds = (const float*)d_in[0];
  const float* w_x    = (const float*)d_in[1];
  const float* w_h    = (const float*)d_in[2];
  const float* b_i    = (const float*)d_in[3];
  const float* b_f    = (const float*)d_in[4];
  const float* b_o    = (const float*)d_in[5];
  const float* b_g    = (const float*)d_in[6];
  float* out = (float*)d_out;

  char* ws = (char*)d_ws;
  const size_t WBYTES = (size_t)4 * HDIM * EDIM * sizeof(bf16_t); // 8 MB
  bf16_t* wxb = (bf16_t*)(ws);
  bf16_t* whb = (bf16_t*)(ws + WBYTES);
  float*  cst = (float*)(ws + 2 * WBYTES);
  bf16_t* h0  = (bf16_t*)(ws + 2 * WBYTES + (size_t)BATCH * HDIM * sizeof(float));
  bf16_t* h1  = h0 + (size_t)BATCH * HDIM;

  const int wn = 4 * HDIM * EDIM;
  lstm_cvt_weights<<<(wn + 255) / 256, 256, 0, stream>>>(w_x, w_h, wxb, whb, wn);

  const int sn = BATCH * HDIM;
  lstm_init_state<<<(sn + 255) / 256, 256, 0, stream>>>(cst, h0, h1, sn);

  dim3 grid(HDIM / 16);
  for (int t = 0; t < T_STEPS; ++t) {
    const bf16_t* hp = (t & 1) ? h1 : h0;
    bf16_t*       hn = (t & 1) ? h0 : h1;
    lstm_step<<<grid, 128, 0, stream>>>(
        embeds + (size_t)t * BATCH * EDIM, wxb, whb, hp,
        b_i, b_f, b_o, b_g, cst, hn, out + (size_t)t * BATCH * HDIM);
  }
}
